// GAT_71365176590575
// MI455X (gfx1250) — matrix operand order repaired and checked by reading, not compiled, not run on hardware
//
#include <hip/hip_runtime.h>

typedef __attribute__((ext_vector_type(16))) _Float16 v16h;
typedef __attribute__((ext_vector_type(8)))  _Float16 v8h;
typedef __attribute__((ext_vector_type(8)))  float    v8f;
typedef __attribute__((ext_vector_type(4)))  float    v4f;

#define NB    4
#define NN    4096
#define CIN   200
#define HCDIM 128

// ---------------------------------------------------------------------------
// Kernel 1: x = rm @ W^T + bias  (fp32), store x transposed as f16 (xT[b][d][n]),
// plus src = x . a_src, dst = x . a_dst  (fp32, feed the softmax logits).
// One block = 128 threads (thread == output channel d), 16 rows per block.
// ---------------------------------------------------------------------------
__global__ __launch_bounds__(128) void gat_proj(
    const float* __restrict__ rm, const float* __restrict__ W,
    const float* __restrict__ bias, const float* __restrict__ avec,
    _Float16* __restrict__ xT, float* __restrict__ srcv, float* __restrict__ dstv)
{
  __shared__ float Wl[128 * 51];   // 50-wide c-tile of W, stride 51 (conflict-free)
  __shared__ float rml[16 * 51];   // 16 rows x 50 c-tile of rm
  __shared__ float ps[128];
  __shared__ float pt[128];
  const int tid   = threadIdx.x;
  const int rows0 = blockIdx.x * 16;           // 16 consecutive global rows
  const float bv  = bias[tid];
  const float asr = avec[tid];
  const float adr = avec[HCDIM + tid];

  float acc[16];
#pragma unroll
  for (int r = 0; r < 16; ++r) acc[r] = bv;

  for (int ct = 0; ct < 4; ++ct) {             // 4 c-tiles of 50 cover CIN=200
    const int c0 = ct * 50;
    for (int v = tid; v < 128 * 50; v += 128) {
      int d = v / 50, cc = v - d * 50;
      Wl[d * 51 + cc] = W[d * CIN + c0 + cc];
    }
    for (int v = tid; v < 16 * 50; v += 128) {
      int rr = v / 50, cc = v - rr * 50;
      rml[rr * 51 + cc] = rm[(size_t)(rows0 + rr) * CIN + c0 + cc];
    }
    __syncthreads();
    for (int cc = 0; cc < 50; ++cc) {
      float wv = Wl[tid * 51 + cc];            // conflict-free (stride 51)
#pragma unroll
      for (int r = 0; r < 16; ++r) acc[r] += rml[r * 51 + cc] * wv;  // broadcast
    }
    __syncthreads();
  }

  const int b = rows0 >> 12;                   // batch (4096 rows per batch)
#pragma unroll 1
  for (int r = 0; r < 16; ++r) {
    const int row = rows0 + r;
    const int n   = row & (NN - 1);
    xT[((size_t)(b * HCDIM + tid)) * NN + n] = (_Float16)acc[r];
    ps[tid] = acc[r] * asr;
    pt[tid] = acc[r] * adr;
    __syncthreads();
    for (int s2 = 64; s2 > 0; s2 >>= 1) {
      if (tid < s2) { ps[tid] += ps[tid + s2]; pt[tid] += pt[tid + s2]; }
      __syncthreads();
    }
    if (tid == 0) { srcv[row] = ps[0]; dstv[row] = pt[0]; }
    __syncthreads();
  }
}

// ---------------------------------------------------------------------------
// Kernel 2: per-batch max over dst (softmax shift upper bound).
// ---------------------------------------------------------------------------
__global__ __launch_bounds__(256) void gat_maxdst(
    const float* __restrict__ dstv, float* __restrict__ maxdst)
{
  __shared__ float red[256];
  const int b = blockIdx.x, tid = threadIdx.x;
  float m = -3.4e38f;
  for (int j = tid; j < NN; j += 256) m = fmaxf(m, dstv[b * NN + j]);
  red[tid] = m;
  __syncthreads();
  for (int s = 128; s > 0; s >>= 1) {
    if (tid < s) red[tid] = fmaxf(red[tid], red[tid + s]);
    __syncthreads();
  }
  if (tid == 0) maxdst[b] = red[0];
}

// ---------------------------------------------------------------------------
// Kernel 3: fused  e -> leakyrelu -> softmax -> alpha @ x  via WMMA f16.
// Workgroup: 256 threads (8 waves), 16 queries (one M-tile).
// Wave w handles key-blocks jb = w, w+8, ... (32 keys each); cross-wave
// combine of O (16x128 f32) and l (16) through LDS float atomics.
// ---------------------------------------------------------------------------
__global__ __launch_bounds__(256) void gat_attn(
    const float* __restrict__ adj, const _Float16* __restrict__ xT,
    const float* __restrict__ srcv, const float* __restrict__ dstv,
    const float* __restrict__ maxdst, float* __restrict__ out)
{
  __shared__ float lds_l[16];
  __shared__ float lds_O[16 * 132];            // stride 132 -> atomic adds conflict-free

  const int tid = threadIdx.x;
  for (int v = tid; v < 16 * 132; v += 256) lds_O[v] = 0.f;
  if (tid < 16) lds_l[tid] = 0.f;
  __syncthreads();

  const int wg   = blockIdx.x;                 // 1024 workgroups
  const int b    = wg >> 8;                    // 256 M-tiles per batch
  const int i0   = (wg & 255) << 4;
  const int wave = tid >> 5;
  const int lane = tid & 31;
  const int g    = lane >> 4;                  // lane group (K-half selector)
  const int q    = lane & 15;                  // query row within M-tile

  const float s  = srcv[b * NN + i0 + q];
  const float M  = fmaxf(0.f, s + maxdst[b]);  // upper bound of row logits
  const float* adjRow = adj + ((size_t)(b * NN + i0 + q)) * NN;
  const float* dstb   = dstv + b * NN;
  const _Float16* xTb = xT + (size_t)b * HCDIM * NN;
  const int base0 = g * 16;

  v8f acc[8];
#pragma unroll
  for (int t = 0; t < 8; ++t)
#pragma unroll
    for (int r = 0; r < 8; ++r) acc[t][r] = 0.f;

  float lsum = 0.f;

  for (int jb = wave; jb < NN / 32; jb += 8) { // wave-uniform loop, EXEC all-1s
    const int j0 = jb * 32;

    // ---- Build A fragment: P[q, k] = exp(leakyrelu((s+dst)*adj) - M), f16.
    // A 16x32 layout (matches B): halves i -> K = g*16 + i
    v16h afrag;
#pragma unroll
    for (int c = 0; c < 2; ++c) {
      const int koff = base0 + c * 8;
      v4f a0 = __builtin_nontemporal_load((const v4f*)(adjRow + j0 + koff));
      v4f a1 = __builtin_nontemporal_load((const v4f*)(adjRow + j0 + koff + 4));
      v4f d0 = *(const v4f*)(dstb + j0 + koff);
      v4f d1 = *(const v4f*)(dstb + j0 + koff + 4);
      float av[8] = {a0.x, a0.y, a0.z, a0.w, a1.x, a1.y, a1.z, a1.w};
      float dv[8] = {d0.x, d0.y, d0.z, d0.w, d1.x, d1.y, d1.z, d1.w};
#pragma unroll
      for (int t = 0; t < 8; ++t) {
        float e = (s + dv[t]) * av[t];
        e = fmaxf(e, 0.2f * e);                // leaky relu
        float p = __expf(e - M);               // <= 1, no overflow
        _Float16 ph = (_Float16)p;
        lsum += (float)ph;                     // denominator from rounded p
        afrag[c * 8 + t] = ph;
      }
    }

    // ---- 8 WMMAs against the 8 N-tiles of x (B from transposed f16 x).
    // B 32x16 layout: lane col = q, halves i -> K = g*16 + i  (contiguous in xT)
    const _Float16* xbase = xTb + (size_t)q * NN + j0 + g * 16;
#pragma unroll
    for (int t = 0; t < 8; ++t) {
      const _Float16* p0 = xbase + (size_t)(t * 16) * NN;
      union { v16h v; v8h h[2]; } bu;
      bu.h[0] = *(const v8h*)(p0);
      bu.h[1] = *(const v8h*)(p0 + 8);
      acc[t] = __builtin_amdgcn_wmma_f32_16x16x32_f16(
          false, afrag, false, bu.v, (short)0, acc[t], false, false);
    }
  }

  // ---- Cross-wave combine in LDS.
  atomicAdd(&lds_l[q], lsum);
#pragma unroll
  for (int t = 0; t < 8; ++t)
#pragma unroll
    for (int r = 0; r < 8; ++r)                // D layout: VGPR r -> M = r + 8*g
      atomicAdd(&lds_O[(r + 8 * g) * 132 + t * 16 + q], acc[t][r]);
  __syncthreads();

  // ---- out = O / l  (coalesced stores)
  for (int v = tid; v < 16 * HCDIM; v += 256) {
    const int qq = v >> 7, n = v & (HCDIM - 1);
    out[((size_t)(b * NN + i0 + qq)) * HCDIM + n] = lds_O[qq * 132 + n] / lds_l[qq];
  }
}

// ---------------------------------------------------------------------------
extern "C" void kernel_launch(void* const* d_in, const int* in_sizes, int n_in,
                              void* d_out, int out_size, void* d_ws, size_t ws_size,
                              hipStream_t stream) {
  const float* rm   = (const float*)d_in[0];   // (4, 4096, 200)
  const float* adj  = (const float*)d_in[1];   // (4, 4096, 4096)
  const float* W    = (const float*)d_in[2];   // (128, 200)
  const float* bias = (const float*)d_in[3];   // (128,)
  const float* avec = (const float*)d_in[4];   // (256,)
  float* out = (float*)d_out;                  // (4, 4096, 128) f32

  char* ws = (char*)d_ws;
  _Float16* xT  = (_Float16*)ws;                                   // 4 MiB
  float* srcv   = (float*)(ws + (size_t)NB * HCDIM * NN * 2);      // 64 KiB
  float* dstv   = srcv + NB * NN;                                  // 64 KiB
  float* maxdst = dstv + NB * NN;                                  // 16 B

  gat_proj  <<<NB * NN / 16, 128, 0, stream>>>(rm, W, bias, avec, xT, srcv, dstv);
  gat_maxdst<<<NB,           256, 0, stream>>>(dstv, maxdst);
  gat_attn  <<<NB * NN / 16, 256, 0, stream>>>(adj, xT, srcv, dstv, maxdst, out);
}
